// EquivariantAttention_14611478741511
// MI455X (gfx1250) — matrix operand order repaired and checked by reading, not compiled, not run on hardware
//
#include <hip/hip_runtime.h>
#include <hip/hip_bf16.h>
#include <math.h>
#include <float.h>

// Problem constants (from reference)
#define B_    2
#define N_    2048
#define D_    256
#define H_    8
#define DH_   64
#define NN_   32
#define ROT_  32              // DH/2 dims get rotary
#define QKV_  (3 * H_ * DH_)  // 1536
#define OC_   (H_ * DH_)      // 512
#define SCALE_ 0.125f         // 64^-0.5

typedef __attribute__((ext_vector_type(16))) _Float16 v16h;
typedef __attribute__((ext_vector_type(8)))  float    v8f;

// ---------------------------------------------------------------------------
// Kernel 1: brute-force 32-NN per point. One block per (b,i), 256 threads.
// Iterative argmin with (value, index) tie-break matches jax.lax.top_k order.
// ---------------------------------------------------------------------------
__global__ void knn_kernel(const float* __restrict__ coors,
                           int* __restrict__ nbr_idx,
                           float* __restrict__ nbr_rd) {
    const int bi = blockIdx.x;          // 0 .. B*N-1
    const int b  = bi / N_;
    const int i  = bi % N_;
    const int t  = threadIdx.x;         // 256 threads

    __shared__ float d2[N_];
    __shared__ float red_v[256];
    __shared__ int   red_i[256];

    const float* cb = coors + (size_t)b * N_ * 3;
    const float xi = cb[i * 3 + 0];
    const float yi = cb[i * 3 + 1];
    const float zi = cb[i * 3 + 2];

    for (int s = 0; s < N_ / 256; ++s) {
        int j = t + s * 256;
        float dx = xi - cb[j * 3 + 0];
        float dy = yi - cb[j * 3 + 1];
        float dz = zi - cb[j * 3 + 2];
        d2[j] = dx * dx + dy * dy + dz * dz;
    }
    __syncthreads();

    for (int sel = 0; sel < NN_; ++sel) {
        float bv = FLT_MAX; int bj = N_;
        for (int s = 0; s < N_ / 256; ++s) {
            int j = t + s * 256;
            float v = d2[j];
            if (v < bv || (v == bv && j < bj)) { bv = v; bj = j; }
        }
        red_v[t] = bv; red_i[t] = bj;
        __syncthreads();
        for (int off = 128; off > 0; off >>= 1) {
            if (t < off) {
                float ov = red_v[t + off]; int oi = red_i[t + off];
                if (ov < red_v[t] || (ov == red_v[t] && oi < red_i[t])) {
                    red_v[t] = ov; red_i[t] = oi;
                }
            }
            __syncthreads();
        }
        if (t == 0) {
            int wi = red_i[0];
            nbr_idx[(size_t)bi * NN_ + sel] = wi;
            nbr_rd [(size_t)bi * NN_ + sel] = sqrtf(red_v[0]);
            d2[wi] = FLT_MAX;            // mask winner for next round
        }
        __syncthreads();
    }
}

// ---------------------------------------------------------------------------
// Kernel 2/4: row-major GEMM C = A(MxK) * B(KxN) [+ bias], f32 in/out,
// f16 WMMA with f32 accumulation.
// Each wave computes a 16x64 tile: ONE A fragment feeds FOUR 16x16 WMMAs per
// K-step (K templated, unrolled x4 -> dense v_wmma issue).
// Fragment layouts used here (contiguous k, same order in A and B):
//   A 16x32 f16 : lanes 0-15 row M=l, K=0..15 (2/VGPR);
//                 lanes 16-31 same rows, K=16..31.
//   B 32x16 f16 : lane -> column N; lanes 0-15 K=0..15 (2/VGPR), lanes16-31 K=16..31.
//   C/D 16x16 f32: VGPR r -> M = r + 8*(lane>=16), N = lane&15.
// ---------------------------------------------------------------------------
template <int KT, bool HAS_BIAS>
__global__ void wmma_gemm_kernel(const float* __restrict__ A,
                                 const float* __restrict__ Bm,
                                 const float* __restrict__ bias,
                                 float* __restrict__ C,
                                 int M, int Nn) {
    const int wave = threadIdx.x >> 5;
    const int lane = threadIdx.x & 31;
    const int half = lane >> 4;
    const int l16  = lane & 15;

    const int ntN64 = Nn >> 6;                 // 64-wide N tiles
    const int tile  = blockIdx.x * 8 + wave;
    if (tile >= (M >> 4) * ntN64) return;      // wave-uniform: EXEC all-ones at WMMA
    const int tM  = tile / ntN64;
    const int tN  = tile % ntN64;

    const float* Arow  = A  + (size_t)(tM * 16 + l16) * KT;
    const float* Bbase = Bm + tN * 64 + l16;   // + sub*16 per sub-tile

    v8f acc0 = {}, acc1 = {}, acc2 = {}, acc3 = {};

    #pragma unroll 4
    for (int k0 = 0; k0 < KT; k0 += 32) {
        if (k0 + 32 < KT) __builtin_prefetch(Arow + k0 + 32, 0, 3);  // global_prefetch_b8

        // --- A fragment (shared by all 4 N sub-tiles) ---
        v16h a;
        #pragma unroll
        for (int p = 0; p < 8; ++p) {
            const int ka = k0 + half * 16 + 2 * p;
            a[2 * p]     = (_Float16)Arow[ka];
            a[2 * p + 1] = (_Float16)Arow[ka + 1];
        }

        // --- 4 B fragments + 4 WMMAs ---
        v16h b0, b1, b2, b3;
        #pragma unroll
        for (int p = 0; p < 8; ++p) {
            const int kb = k0 + half * 16 + 2 * p;
            const float* r0 = Bbase + (size_t)kb * Nn;
            const float* r1 = Bbase + (size_t)(kb + 1) * Nn;
            b0[2 * p] = (_Float16)r0[0];  b0[2 * p + 1] = (_Float16)r1[0];
            b1[2 * p] = (_Float16)r0[16]; b1[2 * p + 1] = (_Float16)r1[16];
            b2[2 * p] = (_Float16)r0[32]; b2[2 * p + 1] = (_Float16)r1[32];
            b3[2 * p] = (_Float16)r0[48]; b3[2 * p + 1] = (_Float16)r1[48];
        }
        acc0 = __builtin_amdgcn_wmma_f32_16x16x32_f16(false, a, false, b0, (short)0, acc0, false, false);
        acc1 = __builtin_amdgcn_wmma_f32_16x16x32_f16(false, a, false, b1, (short)0, acc1, false, false);
        acc2 = __builtin_amdgcn_wmma_f32_16x16x32_f16(false, a, false, b2, (short)0, acc2, false, false);
        acc3 = __builtin_amdgcn_wmma_f32_16x16x32_f16(false, a, false, b3, (short)0, acc3, false, false);
    }

    // --- branch-free epilogue ---
    float* Cbase = C + (size_t)(tM * 16 + half * 8) * Nn + tN * 64 + l16;
    #pragma unroll
    for (int sub = 0; sub < 4; ++sub) {
        const v8f acc = (sub == 0) ? acc0 : (sub == 1) ? acc1 : (sub == 2) ? acc2 : acc3;
        const float bv = HAS_BIAS ? bias[tN * 64 + sub * 16 + l16] : 0.0f;
        float* Cc = Cbase + sub * 16;
        #pragma unroll
        for (int r = 0; r < 8; ++r) {
            Cc[(size_t)r * Nn] = acc[r] + bv;
        }
    }
}

// ---------------------------------------------------------------------------
// Kernel 3: fused per-point attention + coordinate path.
// One block (256 threads) per (b,i). q-rotary is identity (t=0 -> cos=1,sin=0).
// ---------------------------------------------------------------------------
__global__ void attn_kernel(const float* __restrict__ qkv,
                            const float* __restrict__ coors,
                            const int*   __restrict__ nbr_idx,
                            const float* __restrict__ nbr_rd,
                            const float* __restrict__ w_c1, const float* __restrict__ b_c1,
                            const float* __restrict__ w_c2, const float* __restrict__ b_c2,
                            const float* __restrict__ w_g,  const float* __restrict__ b_g,
                            const float* __restrict__ ln_g, const float* __restrict__ ln_b,
                            const float* __restrict__ cc,
                            float* __restrict__ att_out,    // (B*N, 512) pre-projection
                            float* __restrict__ coors_out)  // (B*N, 3)
{
    const int bi = blockIdx.x;
    const int b  = bi / N_;
    const int i  = bi % N_;
    const int t  = threadIdx.x;

    __shared__ float q_s[OC_];
    __shared__ int   idx_s[NN_];
    __shared__ float rd_s[NN_];
    __shared__ float cs_s[NN_][16];
    __shared__ float sn_s[NN_][16];
    __shared__ float qk_s[NN_][H_];
    __shared__ float cattn_s[NN_][H_];
    __shared__ float sign_s[NN_][H_];
    __shared__ float attn_s[NN_][H_];
    __shared__ float gj_s[NN_];

    const float* qrow = qkv + (size_t)bi * QKV_;
    for (int s = t; s < OC_; s += 256) q_s[s] = qrow[s];
    if (t < NN_) {
        idx_s[t] = nbr_idx[(size_t)bi * NN_ + t];
        rd_s[t]  = nbr_rd [(size_t)bi * NN_ + t];
    }
    __syncthreads();

    // rotary table: theta_p(j) = rd*100 * 10000^(-2p/32), p = 0..15
    for (int s = t; s < NN_ * 16; s += 256) {
        const int j = s >> 4, p = s & 15;
        const float inv_freq = powf(10000.0f, -(float)(2 * p) / (float)ROT_);
        const float th = rd_s[j] * 100.0f * inv_freq;
        float sv, cv;
        sincosf(th, &sv, &cv);
        cs_s[j][p] = cv;
        sn_s[j][p] = sv;
    }
    __syncthreads();

    // qk[h][j] = <q_h, rot(k_j,h)> * SCALE ; thread (h = t/32, j = t%32)
    {
        const int h = t >> 5, j = t & 31;
        const float* krow = qkv + (size_t)(b * N_ + idx_s[j]) * QKV_ + OC_ + h * DH_;
        const float* qh   = q_s + h * DH_;
        float acc = 0.0f;
        #pragma unroll
        for (int p = 0; p < 16; ++p) {
            const float c = cs_s[j][p], s = sn_s[j][p];
            const float k0 = krow[2 * p], k1 = krow[2 * p + 1];
            acc += qh[2 * p]     * (k0 * c - k1 * s)
                 + qh[2 * p + 1] * (k1 * c + k0 * s);
        }
        #pragma unroll
        for (int d = ROT_; d < DH_; ++d) acc += qh[d] * krow[d];
        qk_s[j][h] = acc * SCALE_;
    }
    __syncthreads();

    // coordinate MLP + gate, one thread per neighbor j
    if (t < NN_) {
        float x[H_];
        #pragma unroll
        for (int h = 0; h < H_; ++h) x[h] = qk_s[t][h];
        float hid[16];
        #pragma unroll
        for (int m = 0; m < 16; ++m) {
            float a = b_c1[m];
            #pragma unroll
            for (int h = 0; h < H_; ++h) a += x[h] * w_c1[h * 16 + m];
            hid[m] = 0.5f * a * (1.0f + erff(a * 0.7071067811865475f)); // exact gelu
        }
        #pragma unroll
        for (int h = 0; h < H_; ++h) {
            float a = b_c2[h];
            #pragma unroll
            for (int m = 0; m < 16; ++m) a += hid[m] * w_c2[m * H_ + h];
            cattn_s[t][h] = a;
            float g = b_g[h];
            #pragma unroll
            for (int hh = 0; hh < H_; ++hh) g += x[hh] * w_g[hh * H_ + h];
            sign_s[t][h] = tanhf(g);
        }
    }
    __syncthreads();

    // softmax over neighbors: coor_attn (t<8) and attn (8<=t<16)
    if (t < H_) {
        float mx = -FLT_MAX;
        for (int j = 0; j < NN_; ++j) mx = fmaxf(mx, cattn_s[j][t]);
        float sum = 0.0f;
        for (int j = 0; j < NN_; ++j) { float e = __expf(cattn_s[j][t] - mx); cattn_s[j][t] = e; sum += e; }
        const float inv = 1.0f / sum;
        for (int j = 0; j < NN_; ++j) cattn_s[j][t] *= inv;
    } else if (t < 2 * H_) {
        const int h = t - H_;
        float mx = -FLT_MAX;
        for (int j = 0; j < NN_; ++j) mx = fmaxf(mx, qk_s[j][h]);
        float sum = 0.0f;
        for (int j = 0; j < NN_; ++j) { float e = __expf(qk_s[j][h] - mx); attn_s[j][h] = e; sum += e; }
        const float inv = 1.0f / sum;
        for (int j = 0; j < NN_; ++j) attn_s[j][h] *= inv;
    }
    __syncthreads();

    // g_j = sum_h coor_attn * sign * coors_combine
    if (t < NN_) {
        float g = 0.0f;
        #pragma unroll
        for (int h = 0; h < H_; ++h) g += cattn_s[t][h] * sign_s[t][h] * cc[h];
        gj_s[t] = g;
    }
    __syncthreads();

    // coors_out[b,i,c] ; note xhat == 0 exactly -> phase = ln_b
    if (t < 3) {
        const float* cb = coors + (size_t)b * N_ * 3;
        const float xhat = 0.0f;                    // (nrm - nrm) / sqrt(1e-5)
        const float phase = ln_g[0] * xhat + ln_b[0];
        float acc = 0.0f;
        for (int j = 0; j < NN_; ++j) {
            const int jj = idx_s[j];
            const float rx = cb[i * 3 + 0] - cb[jj * 3 + 0];
            const float ry = cb[i * 3 + 1] - cb[jj * 3 + 1];
            const float rz = cb[i * 3 + 2] - cb[jj * 3 + 2];
            const float nrm = sqrtf(rx * rx + ry * ry + rz * rz);
            const float comp = (t == 0) ? rx : ((t == 1) ? ry : rz);
            acc += gj_s[j] * phase * (comp / fmaxf(nrm, 1e-8f));
        }
        coors_out[(size_t)bi * 3 + t] = acc;
    }

    // out[h, d] = sum_j attn[j][h] * rot(v_j,h)[d]; each thread does 2 dims
    {
        const int h  = t >> 5;
        const int d0 = (t & 31) * 2;
        float o0 = 0.0f, o1 = 0.0f;
        for (int j = 0; j < NN_; ++j) {
            const float* vrow = qkv + (size_t)(b * N_ + idx_s[j]) * QKV_ + 2 * OC_ + h * DH_;
            const float a  = attn_s[j][h];
            const float v0 = vrow[d0], v1 = vrow[d0 + 1];
            if (d0 < ROT_) {
                const int p = d0 >> 1;
                const float c = cs_s[j][p], s = sn_s[j][p];
                o0 += a * (v0 * c - v1 * s);
                o1 += a * (v1 * c + v0 * s);
            } else {
                o0 += a * v0;
                o1 += a * v1;
            }
        }
        att_out[(size_t)bi * OC_ + h * DH_ + d0]     = o0;
        att_out[(size_t)bi * OC_ + h * DH_ + d0 + 1] = o1;
    }
}

// ---------------------------------------------------------------------------
extern "C" void kernel_launch(void* const* d_in, const int* in_sizes, int n_in,
                              void* d_out, int out_size, void* d_ws, size_t ws_size,
                              hipStream_t stream) {
    const float* feats = (const float*)d_in[0];
    const float* coors = (const float*)d_in[1];
    const float* w_qkv = (const float*)d_in[2];
    const float* w_out = (const float*)d_in[3];
    const float* b_out = (const float*)d_in[4];
    const float* w_c1  = (const float*)d_in[5];
    const float* b_c1  = (const float*)d_in[6];
    const float* w_c2  = (const float*)d_in[7];
    const float* b_c2  = (const float*)d_in[8];
    const float* w_g   = (const float*)d_in[9];
    const float* b_g   = (const float*)d_in[10];
    const float* ln_g  = (const float*)d_in[11];
    const float* ln_b  = (const float*)d_in[12];
    const float* cc    = (const float*)d_in[13];

    float* out       = (float*)d_out;                       // (B,N,256)
    float* coors_out = out + (size_t)B_ * N_ * D_;          // (B,N,3)

    // workspace layout
    char*  ws      = (char*)d_ws;
    int*   nbr_idx = (int*)ws;                              // B*N*NN ints
    float* nbr_rd  = (float*)(ws + (size_t)B_ * N_ * NN_ * sizeof(int));
    float* qkv     = (float*)(ws + (size_t)2 * B_ * N_ * NN_ * sizeof(float));
    float* att     = qkv + (size_t)B_ * N_ * QKV_;          // (B*N, 512)

    const int rows = B_ * N_;                               // 4096

    // 1) 32-NN
    knn_kernel<<<rows, 256, 0, stream>>>(coors, nbr_idx, nbr_rd);

    // 2) QKV projection: (4096x256)*(256x1536) via WMMA (16x64 tile / wave)
    {
        const int waves = (rows / 16) * (QKV_ / 64);        // 256*24 = 6144
        wmma_gemm_kernel<D_, false><<<waves / 8, 256, 0, stream>>>(
            feats, w_qkv, nullptr, qkv, rows, QKV_);
    }

    // 3) fused attention + coordinate path
    attn_kernel<<<rows, 256, 0, stream>>>(qkv, coors, nbr_idx, nbr_rd,
                                          w_c1, b_c1, w_c2, b_c2, w_g, b_g,
                                          ln_g, ln_b, cc, att, coors_out);

    // 4) output projection: (4096x512)*(512x256) + b_out via WMMA
    {
        const int waves = (rows / 16) * (D_ / 64);          // 256*4 = 1024
        wmma_gemm_kernel<OC_, true><<<waves / 8, 256, 0, stream>>>(
            att, w_out, b_out, out, rows, D_);
    }
}
